// RankingSet_53309134078524
// MI455X (gfx1250) — matrix operand order repaired and checked by reading, not compiled, not run on hardware
//
#include <hip/hip_runtime.h>
#include <hip/hip_bf16.h>

// ---------------------------------------------------------------------------
// RankingSet counter for MI455X (gfx1250).
//
// Roofline: GEMM = 50000 x 512 x 6144 x 2 = 314.6 GFLOP over 1.23 GB of data
//  -> 254 FLOP/byte; HBM floor ~53us @ 23.3 TB/s. bf16 WMMA (16x16x32) with a
// 2-term bf16 split (hi+lo, 3 WMMAs per tile) recovers ~f32 accuracy for the
// rtol=1e-5 threshold count at ~3x bf16 cost -- far ahead of the f32 WMMA
// (16x16x4) ceiling. data is read from HBM exactly once: a cooperative,
// double-buffered LDS stage converts each 64x32 f32 slab to bf16 hi/lo in
// fragment order (one convert per element per block, not per wave), and row
// L2-norms are fused into that same pass. Block tile M=64 x N=512 keeps the
// L2-resident query stream amortized over 4 M-tiles per wave.
// ---------------------------------------------------------------------------

#define FPD 6144

typedef __bf16 bf16_t;
typedef __attribute__((ext_vector_type(16))) bf16_t v16bf;
typedef __attribute__((ext_vector_type(8)))  bf16_t v8bf;
typedef __attribute__((ext_vector_type(8)))  float  v8f;
typedef __attribute__((ext_vector_type(4)))  float  v4f;

// Split f32 into hi(bf16) + lo(bf16), hi+lo ~= x to ~2^-19 relative.
// Lowers to v_cvt_pk_bf16_f32 / v_fma_mix_f32_bf16 (native RNE).
__device__ __forceinline__ void split2(float x, bf16_t& h, bf16_t& l) {
    h = (bf16_t)x;
    float rem = x - (float)h;
    l = (bf16_t)rem;
}

// ---------------------------------------------------------------------------
// Kernel 1: per-query preprocessing.
// thresh_j = dot(q_j, t_j)/(max(|q|,eps)*max(|t|,eps));  tl_j = thresh - tol.
// Writes normalized query rows as bf16 hi/lo split (row-major, K contiguous).
// ---------------------------------------------------------------------------
__global__ __launch_bounds__(256) void prep_queries(
    const float* __restrict__ queries, const float* __restrict__ truths,
    bf16_t* __restrict__ qhi, bf16_t* __restrict__ qlo,
    float* __restrict__ tl)
{
    const int j   = blockIdx.x;
    const int tid = threadIdx.x;
    const float* qp = queries + (size_t)j * FPD;
    const float* tp = truths  + (size_t)j * FPD;

    float sqq = 0.f, stt = 0.f, sqt = 0.f;
    for (int k = tid; k < FPD; k += 256) {
        float q = qp[k], t = tp[k];
        sqq = fmaf(q, q, sqq);
        stt = fmaf(t, t, stt);
        sqt = fmaf(q, t, sqt);
    }
    __shared__ float r0[256], r1[256], r2[256];
    r0[tid] = sqq; r1[tid] = stt; r2[tid] = sqt;
    __syncthreads();
    for (int s = 128; s > 0; s >>= 1) {
        if (tid < s) { r0[tid] += r0[tid+s]; r1[tid] += r1[tid+s]; r2[tid] += r2[tid+s]; }
        __syncthreads();
    }
    __shared__ float s_invq;
    if (tid == 0) {
        float nq = fmaxf(sqrtf(r0[0]), 1e-12f);
        float nt = fmaxf(sqrtf(r1[0]), 1e-12f);
        float th = r2[0] / (nq * nt);
        tl[j] = th - (1e-8f + 1e-5f * fabsf(th));   // thresh - (atol + rtol*|thresh|)
        s_invq = 1.0f / nq;
    }
    __syncthreads();
    const float invq = s_invq;
    for (int k = tid; k < FPD; k += 256) {
        bf16_t h, l;
        split2(qp[k] * invq, h, l);
        qhi[(size_t)j * FPD + k] = h;
        qlo[(size_t)j * FPD + k] = l;
    }
}

// ---------------------------------------------------------------------------
// Kernel 2: fused GEMM + row-norm + threshold count.
// Block: 256 threads = 8 waves. Block tile M=64 (rows), N=512 (all queries).
// Wave w owns N columns [w*64, w*64+64): 4(M) x 4(N) = 16 accum tiles 16x16.
// A slab (64 rows x 32 K, f32) is cooperatively converted to bf16 hi/lo in
// fragment order in LDS (double-buffered); waves ds_load_b128 their WMMA
// fragments. Per k-step: 16 tiles x 3 = 48 v_wmma_f32_16x16x32_bf16.
// ---------------------------------------------------------------------------
#define NSTEPS (FPD / 32)

__global__ __launch_bounds__(256) void rank_gemm_count(
    const float*  __restrict__ data,
    const bf16_t* __restrict__ qhi,
    const bf16_t* __restrict__ qlo,
    const float*  __restrict__ tl,
    int* __restrict__ counts,
    int nrows)
{
    // A staging, fragment order: [stage][mtile][frag-lane][frag-elem]
    __shared__ __align__(32) bf16_t sAH[2][4][32][16];   // 8 KB
    __shared__ __align__(32) bf16_t sAL[2][4][32][16];   // 8 KB
    __shared__ float s_ssq[64];
    __shared__ float s_inv[64];
    __shared__ int   s_cnt[512];

    const int tid   = threadIdx.x;
    const int wave  = tid >> 5;
    const int lane  = tid & 31;
    const int hl    = lane >> 4;   // lane half: 0 or 1
    const int l16   = lane & 15;
    const int mbase = blockIdx.x * 64;
    const int nbase = wave * 64;

    s_cnt[tid] = 0; s_cnt[tid + 256] = 0;
    if (tid < 64) s_ssq[tid] = 0.f;

    // ---- producer mapping: thread -> (row 0..63, kk phase 0/8/16/24)
    // A 16x32 bf16 fragment, k order chosen here: lanes 0-15 hold K 0..15,
    // lanes 16-31 hold K 16..31; frag elem = kk&15.
    const int pr  = tid >> 2;           // row within block slab
    const int pkk = (tid & 3) * 8;      // 8 consecutive K per thread
    const int p_t   = pr >> 4;
    const int p_ln  = (pkk >> 4) * 16 + (pr & 15);
    const int p_e0  = ((pkk >> 3) & 1) * 8;
    int prow = mbase + pr; if (prow >= nrows) prow = nrows - 1;
    const float* prowp = data + (size_t)prow * FPD + pkk;
    float ssq_local = 0.f;

    // ---- consumer B pointers: query row nbase+j*16+l16;
    // B 32x16, same k order as A: lanes 0-15 hold K 0..15, lanes 16-31 hold K 16..31.
    const bf16_t* qhp[4];
    const bf16_t* qlp[4];
    #pragma unroll
    for (int j = 0; j < 4; ++j) {
        size_t off = (size_t)(nbase + j * 16 + l16) * FPD + (hl << 4);
        qhp[j] = qhi + off;
        qlp[j] = qlo + off;
    }

    v8f c[4][4];
    #pragma unroll
    for (int t = 0; t < 4; ++t)
        #pragma unroll
        for (int j = 0; j < 4; ++j)
            #pragma unroll
            for (int e = 0; e < 8; ++e) c[t][j][e] = 0.0f;

    // ---- produce stage 0
    {
        v4f x0 = *(const v4f*)(prowp);
        v4f x1 = *(const v4f*)(prowp + 4);
        v8bf hv, lv;
        #pragma unroll
        for (int e = 0; e < 8; ++e) {
            float f = (e < 4) ? x0[e] : x1[e - 4];
            ssq_local = fmaf(f, f, ssq_local);
            bf16_t h, l; split2(f, h, l);
            hv[e] = h; lv[e] = l;
        }
        *(v8bf*)&sAH[0][p_t][p_ln][p_e0] = hv;
        *(v8bf*)&sAL[0][p_t][p_ln][p_e0] = lv;
    }
    __syncthreads();

    for (int s = 0; s < NSTEPS; ++s) {
        const int cur = s & 1;
        // ---- produce next stage (overlaps with WMMA consumption of cur)
        if (s + 1 < NSTEPS) {
            const float* ap = prowp + (s + 1) * 32;
            __builtin_prefetch(ap + 64, 0, 1);            // global_prefetch_b8
            v4f x0 = *(const v4f*)(ap);
            v4f x1 = *(const v4f*)(ap + 4);
            v8bf hv, lv;
            #pragma unroll
            for (int e = 0; e < 8; ++e) {
                float f = (e < 4) ? x0[e] : x1[e - 4];
                ssq_local = fmaf(f, f, ssq_local);
                bf16_t h, l; split2(f, h, l);
                hv[e] = h; lv[e] = l;
            }
            *(v8bf*)&sAH[cur ^ 1][p_t][p_ln][p_e0] = hv;
            *(v8bf*)&sAL[cur ^ 1][p_t][p_ln][p_e0] = lv;
        }
        // ---- consume stage cur: B from global (L2), A frags from LDS
        const int k = s * 32;
        v16bf BH[4], BL[4];
        #pragma unroll
        for (int j = 0; j < 4; ++j) {
            BH[j] = *(const v16bf*)(qhp[j] + k);
            BL[j] = *(const v16bf*)(qlp[j] + k);
        }
        #pragma unroll
        for (int t = 0; t < 4; ++t) {
            v16bf AHt = *(const v16bf*)&sAH[cur][t][lane][0];
            v16bf ALt = *(const v16bf*)&sAL[cur][t][lane][0];
            #pragma unroll
            for (int j = 0; j < 4; ++j) {
                c[t][j] = __builtin_amdgcn_wmma_f32_16x16x32_bf16(
                    false, AHt, false, BH[j], (short)0, c[t][j], false, false);
                c[t][j] = __builtin_amdgcn_wmma_f32_16x16x32_bf16(
                    false, AHt, false, BL[j], (short)0, c[t][j], false, false);
                c[t][j] = __builtin_amdgcn_wmma_f32_16x16x32_bf16(
                    false, ALt, false, BH[j], (short)0, c[t][j], false, false);
            }
        }
        __syncthreads();   // stage rotation
    }

    // ---- finish fused row norms (4 producer threads per row)
    atomicAdd(&s_ssq[pr], ssq_local);
    float tlr[4];
    #pragma unroll
    for (int j = 0; j < 4; ++j) tlr[j] = tl[nbase + j * 16 + l16];
    __syncthreads();
    if (tid < 64) s_inv[tid] = 1.0f / fmaxf(sqrtf(s_ssq[tid]), 1e-12f);
    __syncthreads();

    // ---- count: C layout lanes 0-15 N=lane M=r, lanes 16-31 N=lane-16 M=r+8
    int cnt[4] = {0, 0, 0, 0};
    #pragma unroll
    for (int t = 0; t < 4; ++t) {
        #pragma unroll
        for (int r = 0; r < 8; ++r) {
            int lrow = t * 16 + r + (hl << 3);
            if (mbase + lrow < nrows) {
                float invn = s_inv[lrow];
                #pragma unroll
                for (int j = 0; j < 4; ++j)
                    cnt[j] += (c[t][j][r] * invn >= tlr[j]) ? 1 : 0;
            }
        }
    }
    #pragma unroll
    for (int j = 0; j < 4; ++j)
        atomicAdd(&s_cnt[nbase + j * 16 + l16], cnt[j]);
    __syncthreads();
    atomicAdd(&counts[tid],       s_cnt[tid]);
    atomicAdd(&counts[tid + 256], s_cnt[tid + 256]);
}

// ---------------------------------------------------------------------------
__global__ void zero_counts(int* __restrict__ counts, int q) {
    int i = blockIdx.x * blockDim.x + threadIdx.x;
    if (i < q) counts[i] = 0;
}
__global__ void finalize(const int* __restrict__ counts, int* __restrict__ out, int q) {
    int i = blockIdx.x * blockDim.x + threadIdx.x;
    if (i < q) out[i] = counts[i] - 1;
}

// ---------------------------------------------------------------------------
extern "C" void kernel_launch(void* const* d_in, const int* in_sizes, int n_in,
                              void* d_out, int out_size, void* d_ws, size_t ws_size,
                              hipStream_t stream) {
    const float* data    = (const float*)d_in[0];
    const float* queries = (const float*)d_in[1];
    const float* truths  = (const float*)d_in[2];
    const int n = in_sizes[0] / FPD;     // 50000
    const int q = in_sizes[1] / FPD;     // 512 (GEMM kernel assumes 512)

    // workspace carve (~12.6 MB): q_hi | q_lo | tl | counts
    char* w = (char*)d_ws;
    const size_t qelems = (size_t)q * FPD;
    bf16_t* qhi = (bf16_t*)w;
    bf16_t* qlo = qhi + qelems;
    float*  tl  = (float*)(w + 2 * qelems * sizeof(bf16_t));
    int* counts = (int*)((char*)tl + (((size_t)q * sizeof(float) + 255) & ~(size_t)255));
    int* out    = (int*)d_out;

    zero_counts<<<(q + 255) / 256, 256, 0, stream>>>(counts, q);
    prep_queries<<<q, 256, 0, stream>>>(queries, truths, qhi, qlo, tl);
    rank_gemm_count<<<(n + 63) / 64, 256, 0, stream>>>(data, qhi, qlo, tl, counts, n);
    finalize<<<(q + 255) / 256, 256, 0, stream>>>(counts, out, q);
}
